// FlashLinearAttentionNet_8229157339854
// MI455X (gfx1250) — matrix operand order repaired and checked by reading, not compiled, not run on hardware
//
#include <hip/hip_runtime.h>

// ---------------------------------------------------------------------------
// FlashLinearAttentionNet for MI455X (gfx1250), bf16 WMMA path.
// B=64, N=2048, D=128, H=256.
// ---------------------------------------------------------------------------

#define BB 64
#define NN 2048
#define DD 128
#define HH 256
#define BN (BB * NN)          // 131072 rows
#define EPS 1e-5f
#define KVSPLIT 8             // n-dimension split for the kv reduction

typedef __bf16 v16bf __attribute__((ext_vector_type(16)));
typedef __bf16 v8bf  __attribute__((ext_vector_type(8)));
typedef float  v8f   __attribute__((ext_vector_type(8)));

__device__ __forceinline__ __bf16 f2bf(float f) {
    unsigned u = __builtin_bit_cast(unsigned, f);
    unsigned r = (u + 0x7FFFu + ((u >> 16) & 1u)) >> 16;   // RNE
    return __builtin_bit_cast(__bf16, (unsigned short)r);
}

__device__ __forceinline__ unsigned mono_key(float f) {
    unsigned u = __builtin_bit_cast(unsigned, f);
    return (u & 0x80000000u) ? ~u : (u | 0x80000000u);     // order-preserving map
}
__device__ __forceinline__ float mono_dec(unsigned k) {
    unsigned u = (k & 0x80000000u) ? (k & 0x7FFFFFFFu) : ~k;
    return __builtin_bit_cast(float, u);
}

// ---------------------------------------------------------------------------
// WMMA fragment helpers (bf16, 16x16x32, wave32).
//   A 16x32: lane m=L&15; elems 0-15 @ K = kk + hi*16 (contiguous in A row m)
//   B 32x16: lane n=L&15; elems 0-15 @ K = kk + hi*16 (contiguous in Bt row n)
//   C 16x16: VGPR r, lane L -> (row r + hi*8, col L&15)
// ---------------------------------------------------------------------------

__device__ __forceinline__ v16bf frag16(const __bf16* p0, const __bf16* p1) {
    v8bf lo = *(const v8bf*)p0;
    v8bf hi = *(const v8bf*)p1;
    return __builtin_shufflevector(lo, hi, 0,1,2,3,4,5,6,7,8,9,10,11,12,13,14,15);
}

// GEMM strip: C[16x128] += A[16x128] * Bt[128x128]^T  (Bt row-major: rows =
// output columns, cols = K).  Used for all the X @ W.T style GEMMs.
__device__ __forceinline__ void gemm_strip(const __bf16* __restrict__ A, int lda,
                                           const __bf16* __restrict__ Bt, int ldb,
                                           v8f acc[8], int lane) {
    const int m  = lane & 15;
    const int hi = lane >> 4;
#pragma unroll
    for (int kk = 0; kk < DD; kk += 32) {
        const __bf16* ap = A + (long)m * lda + kk + hi * 16;
        v16bf a = frag16(ap, ap + 8);
#pragma unroll
        for (int t = 0; t < 8; ++t) {
            const __bf16* bp = Bt + (long)(t * 16 + m) * ldb + kk + hi * 16;
            v16bf b = frag16(bp, bp + 8);
            acc[t] = __builtin_amdgcn_wmma_f32_16x16x32_bf16(
                false, a, false, b, (short)0, acc[t], false, false);
        }
    }
}

// Transpose a 16x128 strip held in LDS (row-major [16 n][128 d], bf16) out to
// global dstT[d][n0..n0+16).  Each lane owns 4 d-columns -> 2x b128 stores.
__device__ __forceinline__ void store_strip_T(const __bf16* __restrict__ s,
                                              __bf16* __restrict__ dstT,
                                              long n0, int lane) {
#pragma unroll
    for (int i = 0; i < 4; ++i) {
        const int d = lane * 4 + i;                  // 0..127
        __align__(16) __bf16 tmp[16];
#pragma unroll
        for (int r = 0; r < 16; ++r) tmp[r] = s[r * DD + d];
        uint4* dp = (uint4*)(dstT + (long)d * NN + n0);
        dp[0] = *(const uint4*)&tmp[0];
        dp[1] = *(const uint4*)&tmp[8];
    }
}

// ---------------------------------------------------------------------------
// Kernel 0 helpers: fp32 -> bf16 convert (grid-stride) and u32 zero-fill.
// ---------------------------------------------------------------------------
__global__ void k_cvt(const float* __restrict__ src, __bf16* __restrict__ dst, long n) {
    for (long i = (long)blockIdx.x * blockDim.x + threadIdx.x; i < n;
         i += (long)gridDim.x * blockDim.x)
        dst[i] = f2bf(src[i]);
}
__global__ void k_zero(unsigned* __restrict__ p, long n) {
    for (long i = (long)blockIdx.x * blockDim.x + threadIdx.x; i < n;
         i += (long)gridDim.x * blockDim.x)
        p[i] = 0u;
}

// ---------------------------------------------------------------------------
// Kernel 1: q/k/v projections + Hadamard feature maps.
// Block = 256 threads = 8 waves; each wave owns a 16-row strip (full D=128).
// qh is stored row-major (A operand of kernel 3); kh and v are stored
// TRANSPOSED per batch (khT[b][d][n], vT[b][e][n]) so the kv reduction is a
// fully contiguous TT GEMM with zero LDS traffic.
// ---------------------------------------------------------------------------
__global__ __launch_bounds__(256) void k_qkvfeat(
    const __bf16* __restrict__ xbf,
    const __bf16* __restrict__ wq, const __bf16* __restrict__ wk,
    const __bf16* __restrict__ wv,
    const __bf16* __restrict__ fq1, const __bf16* __restrict__ fq2,
    const __bf16* __restrict__ fk1, const __bf16* __restrict__ fk2,
    const float* __restrict__ fq1_b, const float* __restrict__ fq2_b,
    const float* __restrict__ fk1_b, const float* __restrict__ fk2_b,
    __bf16* __restrict__ qh, __bf16* __restrict__ khT, __bf16* __restrict__ vT) {
    __shared__ __bf16 lds[8 * 16 * DD];                 // 4 KB per wave
    const int lane = threadIdx.x & 31;
    const int wave = threadIdx.x >> 5;
    const int n = lane & 15, hi = lane >> 4;
    const long row = (long)blockIdx.x * 128 + wave * 16;
    const int b = (int)(row >> 11);                     // batch
    const long n0 = row & (NN - 1);                     // seq offset in batch
    const __bf16* xrow = xbf + row * DD;
    __bf16* myl = &lds[wave * 16 * DD];
    const v8f z = {};

    // ---- q path ----
    v8f q[8];
#pragma unroll
    for (int t = 0; t < 8; ++t) q[t] = z;
    gemm_strip(xrow, DD, wq, DD, q, lane);
#pragma unroll
    for (int t = 0; t < 8; ++t)
#pragma unroll
        for (int r = 0; r < 8; ++r) myl[(r + hi * 8) * DD + t * 16 + n] = f2bf(q[t][r]);

    v8f q1[8], q2[8];
#pragma unroll
    for (int t = 0; t < 8; ++t) { q1[t] = z; q2[t] = z; }
    gemm_strip(myl, DD, fq1, DD, q1, lane);
    gemm_strip(myl, DD, fq2, DD, q2, lane);
#pragma unroll
    for (int t = 0; t < 8; ++t) {
        float b1 = fq1_b[t * 16 + n], b2 = fq2_b[t * 16 + n];
#pragma unroll
        for (int r = 0; r < 8; ++r) q1[t][r] = (q1[t][r] + b1) * (q2[t][r] + b2);
    }
    {
        __bf16* qp = qh + row * DD;
#pragma unroll
        for (int t = 0; t < 8; ++t)
#pragma unroll
            for (int r = 0; r < 8; ++r) qp[(long)(r + hi * 8) * DD + t * 16 + n] = f2bf(q1[t][r]);
    }

    // ---- k path ----
    v8f k[8];
#pragma unroll
    for (int t = 0; t < 8; ++t) k[t] = z;
    gemm_strip(xrow, DD, wk, DD, k, lane);
#pragma unroll
    for (int t = 0; t < 8; ++t)
#pragma unroll
        for (int r = 0; r < 8; ++r) myl[(r + hi * 8) * DD + t * 16 + n] = f2bf(k[t][r]);

    v8f k1[8], k2[8];
#pragma unroll
    for (int t = 0; t < 8; ++t) { k1[t] = z; k2[t] = z; }
    gemm_strip(myl, DD, fk1, DD, k1, lane);
    gemm_strip(myl, DD, fk2, DD, k2, lane);
#pragma unroll
    for (int t = 0; t < 8; ++t) {
        float b1 = fk1_b[t * 16 + n], b2 = fk2_b[t * 16 + n];
#pragma unroll
        for (int r = 0; r < 8; ++r) k1[t][r] = (k1[t][r] + b1) * (k2[t][r] + b2);
    }
    // stage final k in LDS, then transpose-store to khT[b][d][n]
#pragma unroll
    for (int t = 0; t < 8; ++t)
#pragma unroll
        for (int r = 0; r < 8; ++r) myl[(r + hi * 8) * DD + t * 16 + n] = f2bf(k1[t][r]);
    store_strip_T(myl, khT + (long)b * DD * NN, n0, lane);

    // ---- v path ----
    v8f v[8];
#pragma unroll
    for (int t = 0; t < 8; ++t) v[t] = z;
    gemm_strip(xrow, DD, wv, DD, v, lane);
#pragma unroll
    for (int t = 0; t < 8; ++t)
#pragma unroll
        for (int r = 0; r < 8; ++r) myl[(r + hi * 8) * DD + t * 16 + n] = f2bf(v[t][r]);
    store_strip_T(myl, vT + (long)b * DD * NN, n0, lane);
}

// ---------------------------------------------------------------------------
// Kernel 2: kv[b][d][e] = sum_n k[b][n][d] v[b][n][e], as a contiguous TT GEMM
// over khT[b][d][n] x vT[b][e][n].  Grid = BB*KVSPLIT blocks; each block
// reduces a 256-long n-chunk and accumulates into kvAcc (f32, [b][e][d],
// pre-transposed for the q@kv GEMM) via global_atomic_add_f32.
// ---------------------------------------------------------------------------
__global__ __launch_bounds__(256) void k_kv(const __bf16* __restrict__ khT,
                                            const __bf16* __restrict__ vT,
                                            float* __restrict__ kvAcc) {
    const int b = blockIdx.x / KVSPLIT;
    const int chunk = blockIdx.x % KVSPLIT;
    const int lane = threadIdx.x & 31;
    const int wave = threadIdx.x >> 5;
    const int m = lane & 15, hi = lane >> 4;
    const int drow = wave * 16;
    const __bf16* ka = khT + (long)b * DD * NN;  // [d][n]
    const __bf16* va = vT + (long)b * DD * NN;   // [e][n]
    const v8f z = {};
    v8f acc[8];
#pragma unroll
    for (int t = 0; t < 8; ++t) acc[t] = z;

    const int nbeg = chunk * (NN / KVSPLIT);
    const int nend = nbeg + (NN / KVSPLIT);
    for (int nk = nbeg; nk < nend; nk += 32) {
        const __bf16* ap = ka + (long)(drow + m) * NN + nk + hi * 16;
        __builtin_prefetch(ap + 32, 0, 0);       // next n-chunk of this A row
        v16bf a = frag16(ap, ap + 8);
#pragma unroll
        for (int t = 0; t < 8; ++t) {
            const __bf16* bp = va + (long)(t * 16 + m) * NN + nk + hi * 16;
            v16bf bf = frag16(bp, bp + 8);
            acc[t] = __builtin_amdgcn_wmma_f32_16x16x32_bf16(
                false, a, false, bf, (short)0, acc[t], false, false);
        }
    }

    float* kvb = kvAcc + (long)b * DD * DD;      // [e][d]
#pragma unroll
    for (int t = 0; t < 8; ++t) {
        int e = t * 16 + m;
#pragma unroll
        for (int r = 0; r < 8; ++r)
            atomicAdd(&kvb[(long)e * DD + drow + r + hi * 8], acc[t][r]);
    }
}

// ---------------------------------------------------------------------------
// Kernel 3: o = q @ kv, RMSNorm over D, o @ Wo^T, monotone-key max-pool.
// ---------------------------------------------------------------------------
__global__ __launch_bounds__(256) void k_out(const __bf16* __restrict__ qh,
                                             const __bf16* __restrict__ kvT,
                                             const __bf16* __restrict__ wo,
                                             const float* __restrict__ norm_w,
                                             unsigned* __restrict__ pooledKey) {
    __shared__ __bf16 lds[8 * 16 * DD];
    const int lane = threadIdx.x & 31;
    const int wave = threadIdx.x >> 5;
    const int n = lane & 15, hi = lane >> 4;
    const long row = (long)blockIdx.x * 128 + wave * 16;
    const int b = (int)(row >> 11);              // row / 2048
    const __bf16* qrow = qh + row * DD;
    const __bf16* kvb = kvT + (long)b * DD * DD; // row-major [e][d]
    __bf16* myl = &lds[wave * 16 * DD];
    const v8f z = {};

    v8f o[8];
#pragma unroll
    for (int t = 0; t < 8; ++t) o[t] = z;
    gemm_strip(qrow, DD, kvb, DD, o, lane);

    // RMSNorm over the 128-wide rows, done on C-fragment layout.
    float part[8];
#pragma unroll
    for (int r = 0; r < 8; ++r) {
        float s = 0.f;
#pragma unroll
        for (int t = 0; t < 8; ++t) s += o[t][r] * o[t][r];
        part[r] = s;
    }
#pragma unroll
    for (int r = 0; r < 8; ++r) {
        part[r] += __shfl_xor(part[r], 1, 32);
        part[r] += __shfl_xor(part[r], 2, 32);
        part[r] += __shfl_xor(part[r], 4, 32);
        part[r] += __shfl_xor(part[r], 8, 32);
        part[r] = rsqrtf(part[r] * (1.f / DD) + EPS);
    }
#pragma unroll
    for (int t = 0; t < 8; ++t) {
        float nw = norm_w[t * 16 + n];
#pragma unroll
        for (int r = 0; r < 8; ++r) o[t][r] *= part[r] * nw;
    }

    // stage normalized strip in LDS (bf16) -> A-fragments for the Wo GEMM
#pragma unroll
    for (int t = 0; t < 8; ++t)
#pragma unroll
        for (int r = 0; r < 8; ++r) myl[(r + hi * 8) * DD + t * 16 + n] = f2bf(o[t][r]);

    v8f o2[8];
#pragma unroll
    for (int t = 0; t < 8; ++t) o2[t] = z;
    gemm_strip(myl, DD, wo, DD, o2, lane);

    // max-pool over sequence (rows): local reduce over 8 rows then one atomic
#pragma unroll
    for (int t = 0; t < 8; ++t) {
        unsigned mk = 0u;
#pragma unroll
        for (int r = 0; r < 8; ++r) {
            unsigned kk = mono_key(o2[t][r]);
            mk = (kk > mk) ? kk : mk;
        }
        atomicMax(&pooledKey[b * DD + t * 16 + n], mk);
    }
}

// ---------------------------------------------------------------------------
// Kernel 4: out[b,h] = pooled[b,:] . proj_w[h,:] + proj_b[h]   (tiny)
// ---------------------------------------------------------------------------
__global__ __launch_bounds__(256) void k_proj(const unsigned* __restrict__ pooledKey,
                                              const float* __restrict__ proj_w,
                                              const float* __restrict__ proj_b,
                                              float* __restrict__ out) {
    __shared__ float p[DD];
    const int b = blockIdx.x, t = threadIdx.x;
    if (t < DD) p[t] = mono_dec(pooledKey[b * DD + t]);
    __syncthreads();
    float s = proj_b[t];
    const float* w = proj_w + (long)t * DD;
#pragma unroll 8
    for (int d = 0; d < DD; ++d) s += p[d] * w[d];
    out[(long)b * HH + t] = s;
}

// ---------------------------------------------------------------------------
extern "C" void kernel_launch(void* const* d_in, const int* in_sizes, int n_in,
                              void* d_out, int out_size, void* d_ws, size_t ws_size,
                              hipStream_t stream) {
    const float* x      = (const float*)d_in[0];
    const float* Wq     = (const float*)d_in[1];
    const float* Wk     = (const float*)d_in[2];
    const float* Wv     = (const float*)d_in[3];
    const float* Wo     = (const float*)d_in[4];
    const float* fq1_w  = (const float*)d_in[5];
    const float* fq1_b  = (const float*)d_in[6];
    const float* fq2_w  = (const float*)d_in[7];
    const float* fq2_b  = (const float*)d_in[8];
    const float* fk1_w  = (const float*)d_in[9];
    const float* fk1_b  = (const float*)d_in[10];
    const float* fk2_w  = (const float*)d_in[11];
    const float* fk2_b  = (const float*)d_in[12];
    const float* norm_w = (const float*)d_in[13];
    const float* proj_w = (const float*)d_in[14];
    const float* proj_b = (const float*)d_in[15];

    const long NX = (long)BN * DD;       // 16,777,216
    const long NW = (long)DD * DD;       // 16,384
    const long NKV = (long)BB * DD * DD; // 1,048,576
    char* ws = (char*)d_ws;
    size_t off = 0;
    __bf16* xbf   = (__bf16*)(ws + off); off += (size_t)NX * 2;   // 32 MiB
    __bf16* qh    = (__bf16*)(ws + off); off += (size_t)NX * 2;
    __bf16* khT   = (__bf16*)(ws + off); off += (size_t)NX * 2;
    __bf16* vT    = (__bf16*)(ws + off); off += (size_t)NX * 2;
    float*  kvAcc = (float*)(ws + off);  off += (size_t)NKV * 4;  // 4 MiB
    __bf16* kvT   = (__bf16*)(ws + off); off += (size_t)NKV * 2;
    __bf16* wbf   = (__bf16*)(ws + off); off += (size_t)8 * NW * 2;
    unsigned* pooledKey = (unsigned*)(ws + off); off += (size_t)BB * DD * 4;
    if (ws_size < off) return;  // workspace too small; bail deterministically

    __bf16* wq_b  = wbf + 0 * NW;
    __bf16* wk_b  = wbf + 1 * NW;
    __bf16* wv_b  = wbf + 2 * NW;
    __bf16* wo_b  = wbf + 3 * NW;
    __bf16* fq1b_ = wbf + 4 * NW;
    __bf16* fq2b_ = wbf + 5 * NW;
    __bf16* fk1b_ = wbf + 6 * NW;
    __bf16* fk2b_ = wbf + 7 * NW;

    // Stage 0: converts + init
    k_cvt<<<4096, 256, 0, stream>>>(x, xbf, NX);
    k_cvt<<<64, 256, 0, stream>>>(Wq, wq_b, NW);
    k_cvt<<<64, 256, 0, stream>>>(Wk, wk_b, NW);
    k_cvt<<<64, 256, 0, stream>>>(Wv, wv_b, NW);
    k_cvt<<<64, 256, 0, stream>>>(Wo, wo_b, NW);
    k_cvt<<<64, 256, 0, stream>>>(fq1_w, fq1b_, NW);
    k_cvt<<<64, 256, 0, stream>>>(fq2_w, fq2b_, NW);
    k_cvt<<<64, 256, 0, stream>>>(fk1_w, fk1b_, NW);
    k_cvt<<<64, 256, 0, stream>>>(fk2_w, fk2b_, NW);
    k_zero<<<1024, 256, 0, stream>>>((unsigned*)kvAcc, NKV);      // 0u == 0.0f
    k_zero<<<32, 256, 0, stream>>>(pooledKey, (long)BB * DD);

    // Stage 1: q/k/v + Hadamard feature maps (8 waves x 16-row strips/block)
    k_qkvfeat<<<BN / 128, 256, 0, stream>>>(xbf, wq_b, wk_b, wv_b,
                                            fq1b_, fq2b_, fk1b_, fk2b_,
                                            fq1_b, fq2_b, fk1_b, fk2_b,
                                            qh, khT, vT);
    // Stage 2: per-batch kv = k^T v, split 8x over n, f32 atomic accumulate
    k_kv<<<BB * KVSPLIT, 256, 0, stream>>>(khT, vT, kvAcc);
    // Stage 2b: kvAcc f32 -> kvT bf16 (already [b][e][d] layout)
    k_cvt<<<1024, 256, 0, stream>>>(kvAcc, kvT, NKV);
    // Stage 3: o = q kv, RMSNorm, Wo, max-pool
    k_out<<<BN / 128, 256, 0, stream>>>(qh, kvT, wo_b, norm_w, pooledKey);
    // Stage 4: final projection
    k_proj<<<BB, 256, 0, stream>>>(pooledKey, proj_w, proj_b, (float*)d_out);
}
